// DimeReactionNN_1503238553654
// MI455X (gfx1250) — compile-verified, matrix operand order checked by reading, not run on hardware
//
#include <hip/hip_runtime.h>
#include <math.h>
#include <stdint.h>

#define H 64
#define NN 8000
#define NE 12000
#define NA 48000
#define NB 64
#define DHO 4096
#define NODE_IN 110
#define NSTEPS 4
#define NPOOL 3
#define D2 128   /* 2*H  */
#define QS 256   /* 2*D2 */

#define KCHUNKS 34         /* 17 k-blocks of 64 -> 34 chunks of 32 */
#define BP_KRON (KCHUNKS * 2048)      /* packed B halves per kron weight set */
#define BP_SP   (64 * 8 * 2048)       /* packed spW halves */

typedef __attribute__((ext_vector_type(16))) _Float16 v16h;
typedef __attribute__((ext_vector_type(8)))  _Float16 v8h;
typedef __attribute__((ext_vector_type(8)))  float    v8f;

__device__ inline v8f wmma16(v16h a, v16h b, v8f c) {
  return __builtin_amdgcn_wmma_f32_16x16x32_f16(false, a, false, b, (short)0, c, false, false);
}

__device__ inline unsigned fkey(float x) {
  unsigned u = __float_as_uint(x);
  return (u & 0x80000000u) ? ~u : (u | 0x80000000u);
}
__device__ inline float funkey(unsigned k) {
  return (k & 0x80000000u) ? __uint_as_float(k ^ 0x80000000u) : __uint_as_float(~k);
}
__device__ inline float sigm(float x) { return 1.f / (1.f + expf(-x)); }

// fragment half index -> k within a 32-chunk (16-bit A/B WMMA layout)
__device__ inline int frag_k(int h, int hi) { return ((h < 8) ? h : h + 8) + hi * 8; }

// ---------------- B pre-pack for kron-GEMM (fragment-major f16) -----------
// layout: bp[kc][slot(=wave*32+lane)][h], kc in [0,34)
__global__ void k_pack_w(const float* __restrict__ W, const float* __restrict__ bias,
                         _Float16* __restrict__ bp) {
  int idx = blockIdx.x * blockDim.x + threadIdx.x;
  if (idx >= BP_KRON) return;
  int kc = idx >> 11;
  int s  = (idx >> 4) & 127;
  int h  = idx & 15;
  int w = s >> 5, L = s & 31;
  int n = (L & 15) + w * 16, hi = L >> 4;
  int k = frag_k(h, hi);
  int kb = kc >> 1;
  int i = (kc & 1) * 32 + k;
  float v = (kb < 16) ? W[kb * 4096 + i * 64 + n] : bias[i * 64 + n];
  bp[idx] = (_Float16)v;
}

// ---------------- B pre-pack for final projection -------------------------
// layout: bp[nblock][kc][slot][h], nblock in [0,64), kc in [0,8)
__global__ void k_pack_sp(const float* __restrict__ W, _Float16* __restrict__ bp) {
  int idx = blockIdx.x * blockDim.x + threadIdx.x;
  if (idx >= BP_SP) return;
  int nb  = idx >> 14;
  int rem = idx & 16383;
  int kc  = rem >> 11;
  int s   = (rem >> 4) & 127;
  int h   = rem & 15;
  int w = s >> 5, L = s & 31;
  int n = (L & 15) + w * 16, hi = L >> 4;
  int k = frag_k(h, hi);
  bp[idx] = (_Float16)W[(kc * 32 + k) * DHO + nb * 64 + n];
}

// ---------------- node projection: x0 = relu(nodeattr @ projW + projb) ----
__global__ void k_node_proj(const float* __restrict__ na, const float* __restrict__ W,
                            const float* __restrict__ b, float* __restrict__ x0,
                            float* __restrict__ nf) {
  __shared__ float sx[NODE_IN];
  int n = blockIdx.x, t = threadIdx.x;            // 64 threads
  for (int i = t; i < NODE_IN; i += 64) sx[i] = na[n * NODE_IN + i];
  __syncthreads();
  float acc = b[t];
  for (int i = 0; i < NODE_IN; ++i) acc += sx[i] * W[i * H + t];
  acc = fmaxf(acc, 0.f);
  x0[n * H + t] = acc;
  nf[n * H + t] = acc;
}

// ---------------- edge / angle features -----------------------------------
__global__ void k_edge_feat(const float* __restrict__ ea, const float* __restrict__ el,
                            float* __restrict__ len_rbf, float* __restrict__ ef) {
  int idx = blockIdx.x * blockDim.x + threadIdx.x;
  if (idx >= NE * 8) return;
  int e = idx >> 3, k = idx & 7;
  const float step = 4.0f / 7.0f;
  const float gamma = 1.0f / (step * step);
  float d = el[e] - (float)k * step;
  float v = expf(-gamma * d * d);
  len_rbf[e * 8 + k] = v;
  ef[e * 16 + k] = ea[e * 8 + k];
  ef[e * 16 + 8 + k] = v;
}

__global__ void k_angle_feat(const float* __restrict__ ang, const int* __restrict__ asrc,
                             const float* __restrict__ len_rbf, float* __restrict__ eef) {
  int idx = blockIdx.x * blockDim.x + threadIdx.x;
  if (idx >= NA * 8) return;
  int a = idx >> 3, k = idx & 7;
  const float step = 3.14159265358979323846f / 7.0f;
  const float gamma = 1.0f / (step * step);
  float d = ang[a] - (float)k * step;
  eef[a * 16 + k] = expf(-gamma * d * d);
  eef[a * 16 + 8 + k] = len_rbf[asrc[a] * 8 + k];
}

// ---------------- per-bond updates ----------------------------------------
__global__ void k_bond_update(const float* __restrict__ nf, const int* __restrict__ esrc,
                              const float* __restrict__ auW, const float* __restrict__ aub,
                              const float* __restrict__ buW, const float* __restrict__ bub,
                              float* __restrict__ ha, float* __restrict__ hb) {
  __shared__ float sx[H];
  int e = blockIdx.x, t = threadIdx.x;            // 64 threads
  int s = esrc[e];
  sx[t] = nf[s * H + t];
  __syncthreads();
  float a1 = aub[t], a2 = bub[t];
  for (int i = 0; i < H; ++i) {
    float x = sx[i];
    a1 += x * auW[i * H + t];
    a2 += x * buW[i * H + t];
  }
  ha[e * H + t] = fmaxf(a1, 0.f);
  hb[e * H + t] = fmaxf(a2, 0.f);
}

// ---------------- kron-GEMM, packed-B, fused scatter-add ------------------
// out[scat[r], :] += (kron(feat[r], h[gidx?gidx[r]:r]) | bias-block) @ Wr
// Dual accumulators (even/odd 32-chunk) so adjacent WMMAs are independent
// and the hazard window is filled by the other chunk's loads/pk_muls.
__global__ __launch_bounds__(128) void k_kron_gemm(
    const float* __restrict__ feat,   // [M,16]
    const float* __restrict__ hsrc,   // [*,64]
    const int* __restrict__ gidx,     // gather index or nullptr
    const _Float16* __restrict__ bp,  // packed B [34][128][16]
    const int* __restrict__ sidx,     // scatter index [M]
    float* __restrict__ outAcc,       // [*,64] (atomic accum)
    int M) {
  __shared__ _Float16 shg16[16][H];   // gathered h rows (f16), 128B rows
  __shared__ _Float16 cf16[16][18];   // 17 coefs per row (pad to 18)
  __shared__ int scat[16];

  int t = threadIdx.x;
  int lane = t & 31;
  int wave = t >> 5;
  int m0 = blockIdx.x * 16;

  for (int idx = t; idx < 16 * H; idx += 128) {
    int r = idx >> 6, i = idx & 63;
    int row = m0 + r;
    float v = 0.f;
    if (row < M) {
      int g = gidx ? gidx[row] : row;
      v = hsrc[g * H + i];
    }
    shg16[r][i] = (_Float16)v;
  }
  for (int idx = t; idx < 16 * 17; idx += 128) {
    int r = idx / 17, k = idx % 17;
    int row = m0 + r;
    cf16[r][k] = (_Float16)((row < M) ? ((k < 16) ? feat[row * 16 + k] : 1.0f) : 0.f);
  }
  if (t < 16) scat[t] = (m0 + t < M) ? sidx[m0 + t] : 0;
  __syncthreads();

  v8f acc0 = {}, acc1 = {};
  int m = lane & 15, hi = lane >> 4;
  const v8h* arow = (const v8h*)&shg16[m][0];     // 8 x v8h per row
  const v16h* brow = (const v16h*)bp;             // [kc*128 + slot]
  int slot = wave * 32 + lane;

  __builtin_prefetch(bp + slot * 16, 0, 1);        // warm packed B (L2)

  for (int kb = 0; kb < 17; ++kb) {
    _Float16 cfh = cf16[m][kb];
    // chunk 0 (i in [0,32))
    {
      int base = hi;
      v8h lo = arow[base] * cfh;
      v8h hv = arow[base + 2] * cfh;
      v16h af = __builtin_shufflevector(lo, hv, 0, 1, 2, 3, 4, 5, 6, 7,
                                        8, 9, 10, 11, 12, 13, 14, 15);
      v16h bf = brow[(kb * 2 + 0) * 128 + slot];
      acc0 = wmma16(af, bf, acc0);
    }
    // chunk 1 (i in [32,64)) -> independent accumulator
    {
      int base = 4 + hi;
      v8h lo = arow[base] * cfh;
      v8h hv = arow[base + 2] * cfh;
      v16h af = __builtin_shufflevector(lo, hv, 0, 1, 2, 3, 4, 5, 6, 7,
                                        8, 9, 10, 11, 12, 13, 14, 15);
      v16h bf = brow[(kb * 2 + 1) * 128 + slot];
      acc1 = wmma16(af, bf, acc1);
    }
  }

  int n = lane & 15;
#pragma unroll
  for (int r = 0; r < 8; ++r) {
    int mm = r + hi * 8;
    int row = m0 + mm;
    if (row < M) unsafeAtomicAdd(&outAcc[scat[mm] * H + wave * 16 + n],
                                 acc0[r] + acc1[r]);
  }
}

// ---------------- node GRU update -----------------------------------------
__global__ void k_gru(const float* __restrict__ nacc, const float* __restrict__ gnnb,
                      const float* __restrict__ Wih, const float* __restrict__ Whh,
                      const float* __restrict__ bih, const float* __restrict__ bhh,
                      float* __restrict__ nf) {
  __shared__ float sx[H], sh[H], sgi[3 * H], sgh[3 * H];
  int n = blockIdx.x, t = threadIdx.x;   // 192 threads
  if (t < H) {
    sx[t] = fmaxf(nacc[n * H + t] + gnnb[t], 0.f);
    sh[t] = nf[n * H + t];
  }
  __syncthreads();
  float gi = bih[t], gh = bhh[t];
  for (int i = 0; i < H; ++i) {
    gi += sx[i] * Wih[t * H + i];
    gh += sh[i] * Whh[t * H + i];
  }
  sgi[t] = gi; sgh[t] = gh;
  __syncthreads();
  if (t < H) {
    float r = sigm(sgi[t] + sgh[t]);
    float z = sigm(sgi[H + t] + sgh[H + t]);
    float nn2 = tanhf(sgi[2 * H + t] + r * sgh[2 * H + t]);
    nf[n * H + t] = (1.f - z) * nn2 + z * sh[t];
  }
}

// ---------------- misc helpers --------------------------------------------
__global__ void k_fill(float* __restrict__ p, float v, int n) {
  int i = blockIdx.x * blockDim.x + threadIdx.x;
  if (i < n) p[i] = v;
}

__global__ void k_agg(const float* __restrict__ nf, const float* __restrict__ x0,
                      float* __restrict__ agg) {
  int idx = blockIdx.x * blockDim.x + threadIdx.x;
  if (idx >= NN * D2) return;
  int n = idx >> 7, d = idx & 127;
  agg[idx] = (d < H) ? nf[n * H + d] : x0[n * H + (d - H)];
}

// ---------------- Set2Set LSTM (one graph per block, 128 threads) ---------
__global__ __launch_bounds__(128) void k_lstm(const float* __restrict__ x, int xdim,
    const float* __restrict__ Wih, const float* __restrict__ Whh,
    const float* __restrict__ bih, const float* __restrict__ bhh,
    float* __restrict__ h, float* __restrict__ c) {
  __shared__ float sx[QS], sh[D2], sg[4 * D2];
  int b = blockIdx.x, t = threadIdx.x;
  for (int i = t; i < xdim; i += 128) sx[i] = x[b * xdim + i];
  sh[t] = h[b * D2 + t];
  __syncthreads();
#pragma unroll
  for (int j4 = 0; j4 < 4; ++j4) {
    int j = t + j4 * 128;
    float g = bih[j] + bhh[j];
    const float* wi = &Wih[j * xdim];
    for (int i = 0; i < xdim; ++i) g += sx[i] * wi[i];
    const float* wh = &Whh[j * D2];
    for (int i = 0; i < D2; ++i) g += sh[i] * wh[i];
    sg[j] = g;
  }
  __syncthreads();
  float ig = sigm(sg[t]);
  float fg = sigm(sg[D2 + t]);
  float gg = tanhf(sg[2 * D2 + t]);
  float og = sigm(sg[3 * D2 + t]);
  float c2 = fg * c[b * D2 + t] + ig * gg;
  c[b * D2 + t] = c2;
  h[b * D2 + t] = og * tanhf(c2);
}

// ---------------- attention (segment softmax) -----------------------------
__global__ void k_att_init(unsigned* __restrict__ mbuf, float* __restrict__ sbuf,
                           float* __restrict__ ro) {
  int i = blockIdx.x * blockDim.x + threadIdx.x;
  if (i < NB) { mbuf[i] = fkey(-3.0e38f); sbuf[i] = 0.f; }
  if (i < NB * D2) ro[i] = 0.f;
}

__global__ void k_att_e(const float* __restrict__ agg, const float* __restrict__ q,
                        const int* __restrict__ gids, float* __restrict__ ebuf,
                        unsigned* __restrict__ mbuf) {
  int wid = (blockIdx.x * blockDim.x + threadIdx.x) >> 5;   // one wave per node
  int lane = threadIdx.x & 31;
  if (wid >= NN) return;
  int g = gids[wid];
  float s = 0.f;
  for (int d = lane; d < D2; d += 32) s += agg[wid * D2 + d] * q[g * D2 + d];
  for (int off = 16; off; off >>= 1) s += __shfl_xor(s, off, 32);
  if (lane == 0) { ebuf[wid] = s; atomicMax(&mbuf[g], fkey(s)); }
}

__global__ void k_att_ex(const float* __restrict__ ebuf, const unsigned* __restrict__ mbuf,
                         const int* __restrict__ gids, float* __restrict__ exbuf,
                         float* __restrict__ sbuf) {
  int n = blockIdx.x * blockDim.x + threadIdx.x;
  if (n >= NN) return;
  int g = gids[n];
  float ex = expf(ebuf[n] - funkey(mbuf[g]));
  exbuf[n] = ex;
  unsafeAtomicAdd(&sbuf[g], ex);
}

__global__ void k_att_read(const float* __restrict__ agg, const float* __restrict__ exbuf,
                           const float* __restrict__ sbuf, const int* __restrict__ gids,
                           float* __restrict__ ro) {
  int idx = blockIdx.x * blockDim.x + threadIdx.x;
  if (idx >= NN * D2) return;
  int n = idx >> 7, d = idx & 127;
  int g = gids[n];
  float alpha = exbuf[n] / sbuf[g];
  unsafeAtomicAdd(&ro[g * D2 + d], alpha * agg[idx]);
}

__global__ void k_qstar(const float* __restrict__ h1, const float* __restrict__ ro,
                        float* __restrict__ qs) {
  int idx = blockIdx.x * blockDim.x + threadIdx.x;
  if (idx >= NB * QS) return;
  int b = idx >> 8, j = idx & 255;
  qs[idx] = (j < D2) ? h1[b * D2 + j] : ro[b * D2 + (j - D2)];
}

// ---------------- final projection (WMMA, packed B) + PReLU ---------------
__global__ __launch_bounds__(128) void k_final(const float* __restrict__ qs,
    const _Float16* __restrict__ bp, const float* __restrict__ bias,
    const float* __restrict__ pa, float* __restrict__ out) {
  __shared__ _Float16 sA16[16][QS];   // 16 rows x 256 halves, 512B rows
  int t = threadIdx.x, lane = t & 31, wave = t >> 5;
  int m0 = blockIdx.y * 16, nbase = blockIdx.x * 64;
  for (int idx = t; idx < 16 * QS; idx += 128) {
    int r = idx >> 8, k = idx & 255;
    sA16[r][k] = (_Float16)qs[(m0 + r) * QS + k];
  }
  __syncthreads();
  v8f acc0 = {}, acc1 = {};
  int m = lane & 15, hi = lane >> 4;
  const v8h* arow = (const v8h*)&sA16[m][0];      // 32 x v8h
  const v16h* brow = (const v16h*)bp + blockIdx.x * (8 * 128);
  int slot = wave * 32 + lane;
#pragma unroll
  for (int kp = 0; kp < 4; ++kp) {
    {
      int kc = kp * 2;
      int base = kc * 4 + hi;
      v8h lo = arow[base];
      v8h hv = arow[base + 2];
      v16h af = __builtin_shufflevector(lo, hv, 0, 1, 2, 3, 4, 5, 6, 7,
                                        8, 9, 10, 11, 12, 13, 14, 15);
      acc0 = wmma16(af, brow[kc * 128 + slot], acc0);
    }
    {
      int kc = kp * 2 + 1;
      int base = kc * 4 + hi;
      v8h lo = arow[base];
      v8h hv = arow[base + 2];
      v16h af = __builtin_shufflevector(lo, hv, 0, 1, 2, 3, 4, 5, 6, 7,
                                        8, 9, 10, 11, 12, 13, 14, 15);
      acc1 = wmma16(af, brow[kc * 128 + slot], acc1);
    }
  }
  float a = pa[0];
  int n = lane & 15;
#pragma unroll
  for (int r = 0; r < 8; ++r) {
    int mm = r + hi * 8;
    int col = nbase + wave * 16 + n;
    float v = acc0[r] + acc1[r] + bias[col];
    out[(m0 + mm) * DHO + col] = (v >= 0.f) ? v : a * v;
  }
}

// ---------------- host launcher -------------------------------------------
extern "C" void kernel_launch(void* const* d_in, const int* in_sizes, int n_in,
                              void* d_out, int out_size, void* d_ws, size_t ws_size,
                              hipStream_t stream) {
  const float* node_attribute = (const float*)d_in[0];
  const float* edge_attribute = (const float*)d_in[1];
  const float* edge_length    = (const float*)d_in[2];
  const float* angle          = (const float*)d_in[3];
  const int*   edge_src       = (const int*)d_in[4];
  const int*   edge_dst       = (const int*)d_in[5];
  const int*   angle_src      = (const int*)d_in[6];
  const int*   angle_dst      = (const int*)d_in[7];
  const int*   graph_ids      = (const int*)d_in[8];
  const float* projW = (const float*)d_in[9];
  const float* projb = (const float*)d_in[10];
  const float* bondW = (const float*)d_in[11];
  const float* bondb = (const float*)d_in[12];
  const float* efW   = (const float*)d_in[13];
  const float* efb   = (const float*)d_in[14];
  const float* gnn_b = (const float*)d_in[15];
  const float* buW   = (const float*)d_in[16];
  const float* bub   = (const float*)d_in[17];
  const float* auW   = (const float*)d_in[18];
  const float* aub   = (const float*)d_in[19];
  const float* gWih  = (const float*)d_in[20];
  const float* gWhh  = (const float*)d_in[21];
  const float* gbih  = (const float*)d_in[22];
  const float* gbhh  = (const float*)d_in[23];
  const float* sWih0 = (const float*)d_in[24];
  const float* sWhh0 = (const float*)d_in[25];
  const float* sbih0 = (const float*)d_in[26];
  const float* sbhh0 = (const float*)d_in[27];
  const float* sWih1 = (const float*)d_in[28];
  const float* sWhh1 = (const float*)d_in[29];
  const float* sbih1 = (const float*)d_in[30];
  const float* sbhh1 = (const float*)d_in[31];
  const float* spW   = (const float*)d_in[32];
  const float* spb   = (const float*)d_in[33];
  const float* prelu = (const float*)d_in[34];
  float* out = (float*)d_out;

  float* p = (float*)d_ws;
  float* x0   = p; p += NN * H;
  float* nf   = p; p += NN * H;
  float* ha   = p; p += NE * H;
  float* hb   = p; p += NE * H;
  float* nacc = p; p += NN * H;
  float* len_rbf   = p; p += NE * 8;
  float* edge_feat = p; p += NE * 16;
  float* ee_feat   = p; p += NA * 16;
  float* agg   = p; p += NN * D2;
  float* qstar = p; p += NB * QS;
  float* h0 = p; p += NB * D2;
  float* c0 = p; p += NB * D2;
  float* h1 = p; p += NB * D2;
  float* c1 = p; p += NB * D2;
  float* ebuf  = p; p += NN;
  float* exbuf = p; p += NN;
  float* sbuf  = p; p += NB;
  float* robuf = p; p += NB * D2;
  unsigned* mbuf = (unsigned*)p; p += NB;

  uintptr_t up = (uintptr_t)p;
  up = (up + 31) & ~(uintptr_t)31;
  _Float16* bpEF   = (_Float16*)up; up += (uintptr_t)BP_KRON * 2;
  _Float16* bpBond = (_Float16*)up; up += (uintptr_t)BP_KRON * 2;
  _Float16* bpSP   = (_Float16*)up;

  // one-time (per call) weight packs: constant over steps, L2-resident
  k_pack_w<<<(BP_KRON + 255) / 256, 256, 0, stream>>>(efW, efb, bpEF);
  k_pack_w<<<(BP_KRON + 255) / 256, 256, 0, stream>>>(bondW, bondb, bpBond);
  k_pack_sp<<<(BP_SP + 255) / 256, 256, 0, stream>>>(spW, bpSP);

  k_node_proj<<<NN, 64, 0, stream>>>(node_attribute, projW, projb, x0, nf);
  k_edge_feat<<<(NE * 8 + 255) / 256, 256, 0, stream>>>(edge_attribute, edge_length,
                                                        len_rbf, edge_feat);
  k_angle_feat<<<(NA * 8 + 255) / 256, 256, 0, stream>>>(angle, angle_src, len_rbf, ee_feat);

  for (int s = 0; s < NSTEPS; ++s) {
    k_fill<<<(NN * H + 255) / 256, 256, 0, stream>>>(nacc, 0.f, NN * H);
    k_bond_update<<<NE, 64, 0, stream>>>(nf, edge_src, auW, aub, buW, bub, ha, hb);
    // angle messages: hb += segsum over angles of kron(ee_feat, ha[angle_src]) @ efW
    k_kron_gemm<<<(NA + 15) / 16, 128, 0, stream>>>(ee_feat, ha, angle_src, bpEF,
                                                    angle_dst, hb, NA);
    // bond messages: nacc += segsum over edges of kron(edge_feat, hb) @ bondW
    k_kron_gemm<<<(NE + 15) / 16, 128, 0, stream>>>(edge_feat, hb, nullptr, bpBond,
                                                    edge_dst, nacc, NE);
    k_gru<<<NN, 192, 0, stream>>>(nacc, gnn_b, gWih, gWhh, gbih, gbhh, nf);
  }

  k_agg<<<(NN * D2 + 255) / 256, 256, 0, stream>>>(nf, x0, agg);
  k_fill<<<(NB * QS + 255) / 256, 256, 0, stream>>>(qstar, 0.f, NB * QS);
  k_fill<<<(NB * D2 + 255) / 256, 256, 0, stream>>>(h0, 0.f, NB * D2);
  k_fill<<<(NB * D2 + 255) / 256, 256, 0, stream>>>(c0, 0.f, NB * D2);
  k_fill<<<(NB * D2 + 255) / 256, 256, 0, stream>>>(h1, 0.f, NB * D2);
  k_fill<<<(NB * D2 + 255) / 256, 256, 0, stream>>>(c1, 0.f, NB * D2);

  for (int it = 0; it < NPOOL; ++it) {
    k_lstm<<<NB, 128, 0, stream>>>(qstar, QS, sWih0, sWhh0, sbih0, sbhh0, h0, c0);
    k_lstm<<<NB, 128, 0, stream>>>(h0, D2, sWih1, sWhh1, sbih1, sbhh1, h1, c1);
    k_att_init<<<(NB * D2 + 255) / 256, 256, 0, stream>>>(mbuf, sbuf, robuf);
    k_att_e<<<NN / 4, 128, 0, stream>>>(agg, h1, graph_ids, ebuf, mbuf);
    k_att_ex<<<(NN + 255) / 256, 256, 0, stream>>>(ebuf, mbuf, graph_ids, exbuf, sbuf);
    k_att_read<<<(NN * D2 + 255) / 256, 256, 0, stream>>>(agg, exbuf, sbuf, graph_ids, robuf);
    k_qstar<<<(NB * QS + 255) / 256, 256, 0, stream>>>(h1, robuf, qstar);
  }

  k_final<<<dim3(DHO / 64, 64 / 16), 128, 0, stream>>>(qstar, bpSP, spb, prelu, out);
}
